// ScaledDotProductRelativeAttention_23905787969986
// MI455X (gfx1250) — matrix operand order repaired and checked by reading, not compiled, not run on hardware
//
#include <hip/hip_runtime.h>

typedef __attribute__((ext_vector_type(16))) _Float16 v16h;
typedef __attribute__((ext_vector_type(8)))  _Float16 v8h;
typedef __attribute__((ext_vector_type(4)))  _Float16 v4h;
typedef __attribute__((ext_vector_type(8)))  float    v8f;
typedef __attribute__((ext_vector_type(4)))  float    v4f;

#define B_      32
#define NWAY_   20
#define KSHOT_  50
#define H_      256
#define L_      (NWAY_ * KSHOT_ + 1)   // 1001
#define INV_T_  0.0625f                // 1/16 temperature
#define LPAD    72                     // 64 + 8 f16 pad: 144B rows, 16B aligned

// category map, matching reference override order (5>4>3>0>1>2)
__device__ __forceinline__ int cat_of(int i, int j) {
    if (i == L_ - 1) return (j == L_ - 1) ? 5 : 4;
    if (j == L_ - 1) return 3;
    if (i == j)      return 0;
    if (i / KSHOT_ == j / KSHOT_) return 1;
    return 2;
}

__device__ __forceinline__ v8f wmma_f16(v16h a, v16h b, v8f c) {
    return __builtin_amdgcn_wmma_f32_16x16x32_f16(false, a, false, b,
                                                  (short)0, c, false, false);
}

// build a v16h WMMA operand from two contiguous 8-element LDS reads
__device__ __forceinline__ v16h pack16(const _Float16* p0, const _Float16* p1) {
    v8h a0 = *(const v8h*)p0;
    v8h a1 = *(const v8h*)p1;
    v16h r;
#pragma unroll
    for (int e = 0; e < 8; ++e) { r[e] = a0[e]; r[e + 8] = a1[e]; }
    return r;
}

// ---------------------------------------------------------------------------
// Kernel 1: qe[b,i,c] = dot(q[b,i,:], emb[c,:])   (one wave per (b,i) row)
// ---------------------------------------------------------------------------
__global__ void qe_kernel(const float* __restrict__ q,
                          const float* __restrict__ emb,
                          float* __restrict__ qe) {
    int wave = (blockIdx.x * blockDim.x + threadIdx.x) >> 5;   // b*L + i
    int lane = threadIdx.x & 31;
    if (wave >= B_ * L_) return;
    const float* qrow = q + (size_t)wave * H_;
    float qv[8];
#pragma unroll
    for (int r = 0; r < 8; ++r) qv[r] = qrow[lane + 32 * r];
#pragma unroll
    for (int c = 0; c < 6; ++c) {
        const float* erow = emb + c * H_;
        float s = 0.0f;
#pragma unroll
        for (int r = 0; r < 8; ++r) s += qv[r] * erow[lane + 32 * r];
#pragma unroll
        for (int off = 16; off; off >>= 1) s += __shfl_xor(s, off, 32);
        if (lane == 0) qe[wave * 6 + c] = s;
    }
}

// ---------------------------------------------------------------------------
// Kernel 2: logits[b,i,j] = (q.k + qe[b,i,cat(i,j)]) / T
// 8-wave workgroup computes a 64x64 tile; q/k chunks staged in LDS as f16.
// Each wave owns i-subtile (w&3) and two j-subtiles at (w>>2)*32 + {0,16}.
// ---------------------------------------------------------------------------
__global__ void scores_kernel(const float* __restrict__ q,
                              const float* __restrict__ k,
                              const float* __restrict__ qe,
                              float* __restrict__ logits) {
    __shared__ _Float16 qs[64][LPAD];   // [i_local][h_local]
    __shared__ _Float16 ks[64][LPAD];   // [j_local][h_local]

    int blk = blockIdx.x;
    int b   = blk >> 8;                 // / 256
    int rem = blk & 255;
    int i0  = (rem >> 4) * 64;
    int j0  = (rem & 15) * 64;

    int t    = threadIdx.x;
    int lane = t & 31;
    int half = lane >> 4;
    int m    = lane & 15;
    int w    = t >> 5;
    int isub = (w & 3) * 16;
    int jsub = (w >> 2) * 32;

    int lr = t >> 4;                    // staging: 0..15
    int lc = (t & 15) * 4;              // staging: col 0,4,..,60

    v8f acc0 = {}, acc1 = {};

    for (int h0 = 0; h0 < H_; h0 += 64) {
        __syncthreads();                // LDS reuse fence
#pragma unroll
        for (int p = 0; p < 4; ++p) {
            int row = p * 16 + lr;
            int qi  = min(i0 + row, L_ - 1);
            int kj  = min(j0 + row, L_ - 1);
            v4f qv = *(const v4f*)(q + ((size_t)b * L_ + qi) * H_ + h0 + lc);
            v4f kv = *(const v4f*)(k + ((size_t)b * L_ + kj) * H_ + h0 + lc);
            v4h qh, kh;
#pragma unroll
            for (int c = 0; c < 4; ++c) {
                qh[c] = (_Float16)qv[c];
                kh[c] = (_Float16)kv[c];
            }
            *(v4h*)&qs[row][lc] = qh;
            *(v4h*)&ks[row][lc] = kh;
        }
        __syncthreads();
#pragma unroll
        for (int k0 = 0; k0 < 64; k0 += 32) {
            v16h A  = pack16(&qs[isub + m][k0 + 16 * half],
                             &qs[isub + m][k0 + 16 * half + 8]);
            v16h B0 = pack16(&ks[jsub + m][k0 + 16 * half],
                             &ks[jsub + m][k0 + 16 * half + 8]);
            v16h B1 = pack16(&ks[jsub + 16 + m][k0 + 16 * half],
                             &ks[jsub + 16 + m][k0 + 16 * half + 8]);
            acc0 = wmma_f16(A, B0, acc0);
            acc1 = wmma_f16(A, B1, acc1);
        }
    }

    int jA = j0 + jsub + m;
    int jB = jA + 16;
#pragma unroll
    for (int vv = 0; vv < 8; ++vv) {
        int i = i0 + isub + vv + 8 * half;
        if (i < L_) {
            const float* qer = qe + ((size_t)b * L_ + i) * 6;
            size_t base = ((size_t)b * L_ + i) * (size_t)L_;
            if (jA < L_) logits[base + jA] = (acc0[vv] + qer[cat_of(i, jA)]) * INV_T_;
            if (jB < L_) logits[base + jB] = (acc1[vv] + qer[cat_of(i, jB)]) * INV_T_;
        }
    }
}

// ---------------------------------------------------------------------------
// Kernel 3: in-place row softmax over attn region (one block per row)
// ---------------------------------------------------------------------------
__global__ void softmax_kernel(float* __restrict__ attn) {
    int row = blockIdx.x;                 // b*L + i
    float* x = attn + (size_t)row * L_;
    __shared__ float red[8];
    int t = threadIdx.x;

    float mx = -3.4e38f;
    for (int j = t; j < L_; j += 256) mx = fmaxf(mx, x[j]);
#pragma unroll
    for (int off = 16; off; off >>= 1) mx = fmaxf(mx, __shfl_xor(mx, off, 32));
    if ((t & 31) == 0) red[t >> 5] = mx;
    __syncthreads();
    mx = red[0];
#pragma unroll
    for (int wv = 1; wv < 8; ++wv) mx = fmaxf(mx, red[wv]);
    __syncthreads();

    float sum = 0.0f;
    for (int j = t; j < L_; j += 256) {
        float e = __expf(x[j] - mx);
        x[j] = e;
        sum += e;
    }
#pragma unroll
    for (int off = 16; off; off >>= 1) sum += __shfl_xor(sum, off, 32);
    if ((t & 31) == 0) red[t >> 5] = sum;
    __syncthreads();
    sum = 0.0f;
#pragma unroll
    for (int wv = 0; wv < 8; ++wv) sum += red[wv];
    float inv = 1.0f / sum;
    for (int j = t; j < L_; j += 256) x[j] *= inv;
}

// ---------------------------------------------------------------------------
// Kernel 4: out[b,i,h] = attn[b,i,:] . v[b,:,h]
// 8-wave workgroup computes 64(i) x 64(h); per 64-j chunk, attn staged [i][j]
// and v staged TRANSPOSED [h][j] in LDS (f16), so both WMMA operands are
// contiguous ds_load_b128 pairs. Padded j columns are zero-filled.
// ---------------------------------------------------------------------------
__global__ void out_kernel(const float* __restrict__ attn,
                           const float* __restrict__ v,
                           float* __restrict__ out) {
    __shared__ _Float16 as[64][LPAD];   // [i_local][j_local]
    __shared__ _Float16 vs[64][LPAD];   // [h_local][j_local]  (transposed)

    int blk = blockIdx.x;
    int b   = blk >> 6;                 // / 64
    int rem = blk & 63;
    int i0  = (rem >> 2) * 64;
    int h0  = (rem & 3) * 64;

    int t    = threadIdx.x;
    int lane = t & 31;
    int half = lane >> 4;
    int m    = lane & 15;
    int w    = t >> 5;
    int isub = (w & 3) * 16;
    int hsub = (w >> 2) * 32;

    int lr = t >> 4;                    // attn staging: 0..15
    int lc = (t & 15) * 4;              // attn staging: col 0,4,..,60
    int hc = t & 63;                    // v staging: h col 0..63
    int jg = (t >> 6) * 4;              // v staging: j base 0,4,8,12

    v8f acc0 = {}, acc1 = {};

    for (int j0 = 0; j0 < 1024; j0 += 64) {
        __syncthreads();
        // stage attn tile (scalar loads: rows are only 4B aligned)
#pragma unroll
        for (int p = 0; p < 4; ++p) {
            int row = p * 16 + lr;
            int ai  = min(i0 + row, L_ - 1);
            const float* arow = attn + ((size_t)b * L_ + ai) * (size_t)L_ + j0 + lc;
            v4h ah;
            if (j0 + lc + 3 < L_) {
#pragma unroll
                for (int c = 0; c < 4; ++c) ah[c] = (_Float16)arow[c];
            } else {
#pragma unroll
                for (int c = 0; c < 4; ++c)
                    ah[c] = (j0 + lc + c < L_) ? (_Float16)arow[c] : (_Float16)0.0f;
            }
            *(v4h*)&as[row][lc] = ah;
        }
        // stage v transposed: vs[h][j]; coalesced b32 reads along h
#pragma unroll
        for (int p = 0; p < 4; ++p) {
            int jj = p * 16 + jg;
            v4h vh;
#pragma unroll
            for (int c = 0; c < 4; ++c) {
                int j = j0 + jj + c;
                vh[c] = (j < L_) ? (_Float16)v[((size_t)b * L_ + j) * H_ + h0 + hc]
                                 : (_Float16)0.0f;
            }
            *(v4h*)&vs[hc][jj] = vh;
        }
        __syncthreads();
#pragma unroll
        for (int k0 = 0; k0 < 64; k0 += 32) {
            v16h A  = pack16(&as[isub + m][k0 + 16 * half],
                             &as[isub + m][k0 + 16 * half + 8]);
            v16h B0 = pack16(&vs[hsub + m][k0 + 16 * half],
                             &vs[hsub + m][k0 + 16 * half + 8]);
            v16h B1 = pack16(&vs[hsub + 16 + m][k0 + 16 * half],
                             &vs[hsub + 16 + m][k0 + 16 * half + 8]);
            acc0 = wmma_f16(A, B0, acc0);
            acc1 = wmma_f16(A, B1, acc1);
        }
    }

#pragma unroll
    for (int vv = 0; vv < 8; ++vv) {
        int i = i0 + isub + vv + 8 * half;
        if (i < L_) {
            size_t base = ((size_t)b * L_ + i) * H_ + h0;
            out[base + hsub + m]      = acc0[vv];
            out[base + hsub + 16 + m] = acc1[vv];
        }
    }
}

// ---------------------------------------------------------------------------
extern "C" void kernel_launch(void* const* d_in, const int* in_sizes, int n_in,
                              void* d_out, int out_size, void* d_ws, size_t ws_size,
                              hipStream_t stream) {
    const float* q   = (const float*)d_in[0];
    const float* k   = (const float*)d_in[1];
    const float* v   = (const float*)d_in[2];
    const float* emb = (const float*)d_in[3];
    // d_in[4] = N, d_in[5] = K: compile-time constants here.

    float* out  = (float*)d_out;
    float* attn = out + (size_t)B_ * L_ * H_;   // output tuple: (output, attn) flat
    float* qe   = (float*)d_ws;                 // B*L*6 floats = 768 KB scratch

    // 1) qe = q @ emb^T
    {
        int waves  = B_ * L_;
        int blocks = (waves + 3) / 4;
        qe_kernel<<<blocks, 128, 0, stream>>>(q, emb, qe);
    }
    // 2) logits into attn region: 32 batches x 16 x 16 tiles of 64x64
    scores_kernel<<<B_ * 16 * 16, 256, 0, stream>>>(q, k, qe, attn);
    // 3) softmax in place
    softmax_kernel<<<B_ * L_, 256, 0, stream>>>(attn);
    // 4) out = attn @ v: 32 batches x 16(i) x 4(h) tiles of 64x64
    out_kernel<<<B_ * 16 * 4, 256, 0, stream>>>(attn, v, out);
}
